// ConditionalRandomField_53429393162641
// MI455X (gfx1250) — matrix operand order repaired and checked by reading, not compiled, not run on hardware
//
#include <hip/hip_runtime.h>
#include <hip/hip_bf16.h>

typedef __attribute__((ext_vector_type(16))) _Float16 v16h;
typedef __attribute__((ext_vector_type(8)))  float    v8f;

static constexpr int Bsz = 512;
static constexpr int T   = 1024;
static constexpr int K   = 64;
static constexpr float LN2f   = 0.6931471805599453f;
static constexpr float LOG2Ef = 1.4426950408889634f;

// ---------------------------------------------------------------------------
// Numerator: gold-path score per sequence. One wave (32 lanes) per batch.
// ---------------------------------------------------------------------------
__global__ __launch_bounds__(256) void crf_score_kernel(
    const float* __restrict__ emissions, const float* __restrict__ trans,
    const float* __restrict__ start_t, const float* __restrict__ end_t,
    const int* __restrict__ tags, const unsigned char* __restrict__ mask,
    float* __restrict__ num)
{
    const int wave = threadIdx.x >> 5;
    const int lane = threadIdx.x & 31;
    const int b = blockIdx.x * 8 + wave;
    if (b >= Bsz) return;

    const int* tg           = tags + (size_t)b * T;
    const float* em         = emissions + (size_t)b * T * K;
    const unsigned char* mk = mask + (size_t)b * T;

    float s = 0.0f;
    int   cnt = 0;
    for (int t = lane; t < T; t += 32) cnt += mk[t] ? 1 : 0;
    for (int t = lane + 1; t < T; t += 32) {
        if (mk[t]) {
            const int cur = tg[t], prev = tg[t - 1];
            s += em[(size_t)t * K + cur] + trans[prev * K + cur];
        }
    }
    #pragma unroll
    for (int off = 16; off > 0; off >>= 1) {
        s   += __shfl_xor(s, off, 32);
        cnt += __shfl_xor(cnt, off, 32);
    }
    if (lane == 0) {
        const int t0 = tg[0];
        s += em[t0] + start_t[t0];
        s += end_t[tg[cnt - 1]];
        num[b] = s;
    }
}

// ---------------------------------------------------------------------------
// Forward algorithm (log-partition), 16 sequences per block, 4 waves.
// Wave j owns output columns [16j, 16j+16). Per step:
//   p = exp2(alpha*log2e - rowmax*log2e)   (raw v_exp_f32; f16 A fragments)
//   C = p @ exp(trans)                     (2x v_wmma_f32_16x16x32_f16, K=64)
//   alpha' = sel(mask, rowmax + ln2*log2(C) + emit[t], alpha)
// Ping-pong alpha buffers -> ONE barrier per step. B fragments (exp(trans))
// pinned in VGPRs. Emissions for t+1 prefetched during step t.
// ---------------------------------------------------------------------------
__global__ __launch_bounds__(128) void crf_forward_kernel(
    const float* __restrict__ emissions, const float* __restrict__ transitions,
    const float* __restrict__ start_t, const float* __restrict__ end_t,
    const unsigned char* __restrict__ mask, float* __restrict__ den)
{
    __shared__ float albuf[2][16][66];   // even stride: b64-merge + bank spread
    __shared__ __align__(16) unsigned char msk[16][1024];  // mask slab (16 KB)

    const int tid  = threadIdx.x;
    const int lane = tid & 31;
    const int wave = tid >> 5;        // N-tile index 0..3
    const int m    = lane & 15;       // A: batch row; B/D: column-within-tile
    const int hi   = lane >> 4;       // lane half select
    const int b0   = blockIdx.x * 16; // first batch of this group
    const int col  = wave * 16 + m;   // global state column owned in D layout
    const size_t TK = (size_t)T * K;

    // ---- stage mask slab: rows b0..b0+15 are contiguous, copy 16 KB flat ----
    {
        const uint4* gm = (const uint4*)(mask + (size_t)b0 * T);
        uint4* sm4 = (uint4*)(&msk[0][0]);
        for (int i = tid; i < 16 * T / 16; i += 128) sm4[i] = gm[i];
    }

    // ---- init alpha(t=0) = emissions[:,0,:] + start ----
    for (int idx = tid; idx < 16 * K; idx += 128) {
        const int r = idx >> 6, cc = idx & 63;
        albuf[0][r][cc] = emissions[(size_t)(b0 + r) * TK + cc] + start_t[cc];
    }

    // ---- B fragments: E = exp(transitions), columns n = wave*16 + m ----
    // 16-bit B layout (32x16): half i -> K = (i<8 ? i : i+8) + 8*hi, N = lane&15
    v16h bf0, bf1;
    #pragma unroll
    for (int i = 0; i < 16; ++i) {
        const int k = ((i < 8) ? i : (i + 8)) + hi * 8;
        bf0[i] = (_Float16)__expf(transitions[k * K + col]);
        bf1[i] = (_Float16)__expf(transitions[(k + 32) * K + col]);
    }

    // ---- emission prefetch pipeline: e_cur holds step t's values ----
    const float* ebase = emissions + (size_t)(b0 + hi * 8) * TK + col;
    float e_cur[8];
    #pragma unroll
    for (int r = 0; r < 8; ++r) e_cur[r] = ebase[r * TK + K];   // t = 1

    float (*asrc)[66] = albuf[0];
    float (*adst)[66] = albuf[1];

    __syncthreads();

    for (int t = 1; t < T; ++t) {
        // A fragments from LDS alpha row m.
        // 16-bit A layout (16x32): half i -> K = (i<8 ? i : i+8) + 8*hi
        float v0[16], v1[16];
        float mx = -1e30f;
        #pragma unroll
        for (int i = 0; i < 16; ++i) {
            const int k = ((i < 8) ? i : (i + 8)) + hi * 8;
            v0[i] = asrc[m][k];
            v1[i] = asrc[m][k + 32];
            mx = fmaxf(mx, fmaxf(v0[i], v1[i]));
        }
        mx = fmaxf(mx, __shfl_xor(mx, 16, 32));  // lane pair -> full row max

        // previous alpha at this lane's D-layout positions (masked update)
        float oldv[8];
        #pragma unroll
        for (int r = 0; r < 8; ++r) oldv[r] = asrc[r + hi * 8][col];

        // prefetch next step's emissions (clamped; dead value on last iter)
        const size_t tn = (size_t)((t + 1 < T) ? (t + 1) : (T - 1)) * K;
        float e_next[8];
        #pragma unroll
        for (int r = 0; r < 8; ++r) e_next[r] = ebase[r * TK + tn];

        // p = exp2(v*log2e - mx*log2e): pk_fma + raw v_exp_f32 per element
        const float mxl = mx * LOG2Ef;
        v16h af0, af1;
        #pragma unroll
        for (int i = 0; i < 16; ++i) {
            af0[i] = (_Float16)__builtin_amdgcn_exp2f(fmaf(v0[i], LOG2Ef, -mxl));
            af1[i] = (_Float16)__builtin_amdgcn_exp2f(fmaf(v1[i], LOG2Ef, -mxl));
        }

        v8f c = {};
        c = __builtin_amdgcn_wmma_f32_16x16x32_f16(false, af0, false, bf0,
                                                   (short)0, c, false, false);
        c = __builtin_amdgcn_wmma_f32_16x16x32_f16(false, af1, false, bf1,
                                                   (short)0, c, false, false);

        // D layout: element r -> row = r + 8*hi, col = wave*16 + m.
        // Row-max shift guarantees C in [0.9, 70]: raw v_log_f32 is safe.
        #pragma unroll
        for (int r = 0; r < 8; ++r) {
            const int row = r + hi * 8;
            const float mrow = __shfl(mx, row, 32);   // rowmax lives in lane `row`
            const float nv = fmaf(LN2f, __builtin_amdgcn_logf(c[r]),
                                  mrow + e_cur[r]);
            adst[row][col] = msk[row][t] ? nv : oldv[r];  // write ping-pong dst
            e_cur[r] = e_next[r];
        }

        __syncthreads();   // single barrier: dst stores <-> next step's reads

        float (*tmp)[66] = asrc; asrc = adst; adst = tmp;
    }

    // ---- denominator: logsumexp(alpha + end_transitions) per row ----
    // (asrc points at the last-written buffer after the final swap)
    if (tid < 16) {
        float mx = -1e30f;
        #pragma unroll 8
        for (int k = 0; k < K; ++k)
            mx = fmaxf(mx, asrc[tid][k] + end_t[k]);
        float s = 0.0f;
        #pragma unroll 8
        for (int k = 0; k < K; ++k)
            s += __builtin_amdgcn_exp2f(LOG2Ef * (asrc[tid][k] + end_t[k] - mx));
        den[b0 + tid] = mx + LN2f * __builtin_amdgcn_logf(s);
    }
}

// ---------------------------------------------------------------------------
// Final: out = mean(den - num)
// ---------------------------------------------------------------------------
__global__ __launch_bounds__(256) void crf_reduce_kernel(
    const float* __restrict__ den, const float* __restrict__ num,
    float* __restrict__ out)
{
    __shared__ float sm[256];
    float p = 0.0f;
    for (int i = threadIdx.x; i < Bsz; i += 256) p += den[i] - num[i];
    sm[threadIdx.x] = p;
    __syncthreads();
    for (int off = 128; off > 0; off >>= 1) {
        if (threadIdx.x < off) sm[threadIdx.x] += sm[threadIdx.x + off];
        __syncthreads();
    }
    if (threadIdx.x == 0) out[0] = sm[0] / (float)Bsz;
}

extern "C" void kernel_launch(void* const* d_in, const int* in_sizes, int n_in,
                              void* d_out, int out_size, void* d_ws, size_t ws_size,
                              hipStream_t stream) {
    const float*         emissions   = (const float*)d_in[0];
    const float*         transitions = (const float*)d_in[1];
    const float*         start_t     = (const float*)d_in[2];
    const float*         end_t       = (const float*)d_in[3];
    const int*           tags        = (const int*)d_in[4];
    const unsigned char* mask        = (const unsigned char*)d_in[5];
    float*               out         = (float*)d_out;

    float* den = (float*)d_ws;
    float* num = den + Bsz;

    crf_score_kernel<<<Bsz / 8, 256, 0, stream>>>(emissions, transitions,
                                                  start_t, end_t, tags, mask, num);
    crf_forward_kernel<<<Bsz / 16, 128, 0, stream>>>(emissions, transitions,
                                                     start_t, end_t, mask, den);
    crf_reduce_kernel<<<1, 256, 0, stream>>>(den, num, out);
}
